// PairCosineSim_9371618640063
// MI455X (gfx1250) — matrix operand order repaired and checked by reading, not compiled, not run on hardware
//
#include <hip/hip_runtime.h>
#include <cstdint>
#include <cstddef>

// ---------------------------------------------------------------------------
// Pairwise cosine similarity, MI455X (gfx1250, wave32, WMMA).
//   out[b,t,s] = <X[b,t,:], S[b,s,:]> / max(||X[b,t]|| * ||S[b,s]||, 1e-10)
// B=8, T=S=1024, D=2048, fp32.
// GEMM core uses split-precision f16 WMMA: x = hi + lo (f16), dot via
// hi*hi + hi*lo + lo*hi accumulated in f32 -> ~2^-22 relative error.
// ---------------------------------------------------------------------------

typedef __attribute__((ext_vector_type(16))) _Float16 v16h;
typedef __attribute__((ext_vector_type(8)))  _Float16 h8;
typedef __attribute__((ext_vector_type(4)))  _Float16 h4;
typedef __attribute__((ext_vector_type(8)))  float    v8f;
typedef __attribute__((ext_vector_type(4)))  float    f4;

namespace {
constexpr int kB = 8;
constexpr int kT = 1024;
constexpr int kS = 1024;
constexpr int kD = 2048;

constexpr int BM = 128;   // tile rows (t)
constexpr int BN = 64;    // tile cols (s)
constexpr int BK = 32;    // k step (matches 16x16x32 f16 WMMA)
constexpr int LDH = BK + 8;  // LDS row stride in halves (80 B: 16B-aligned, staggered)
}

// -------------------------- row norms --------------------------------------
__global__ void row_norms_kernel(const float* __restrict__ sup,
                                 const float* __restrict__ xh,
                                 float* __restrict__ xnorm,
                                 float* __restrict__ snorm) {
  __shared__ float red[8];
  const int row = blockIdx.x;
  const float* p;
  float* outp;
  if (row < kB * kT) {
    p = xh + (size_t)row * kD;
    outp = xnorm + row;
  } else {
    const int r = row - kB * kT;
    p = sup + (size_t)r * kD;
    outp = snorm + r;
  }
  const int tid = threadIdx.x;
  const f4* p4 = (const f4*)p;            // 2048 floats = 512 float4; 256 thr * 2
  f4 a = p4[tid * 2 + 0];
  f4 b = p4[tid * 2 + 1];
  float s = a[0]*a[0] + a[1]*a[1] + a[2]*a[2] + a[3]*a[3]
          + b[0]*b[0] + b[1]*b[1] + b[2]*b[2] + b[3]*b[3];
  // wave32 reduction
  #pragma unroll
  for (int off = 16; off > 0; off >>= 1) s += __shfl_xor(s, off, 32);
  if ((tid & 31) == 0) red[tid >> 5] = s;
  __syncthreads();
  if (tid == 0) {
    float t = 0.f;
    #pragma unroll
    for (int i = 0; i < 8; ++i) t += red[i];
    *outp = sqrtf(t);
  }
}

// -------------------------- GEMM helpers -----------------------------------
__device__ __forceinline__ void cvt4(f4 v, h4& hi, h4& lo) {
  #pragma unroll
  for (int i = 0; i < 4; ++i) {
    _Float16 h = (_Float16)v[i];
    hi[i] = h;
    lo[i] = (_Float16)(v[i] - (float)h);
  }
}

// A fragment (16x32 f16, M x K): lane = half*16 + m.
// lanes 0-15: elems 0..7 -> K=0..7,  elems 8..15 -> K=16..23
// lanes 16-31: elems 0..7 -> K=8..15, elems 8..15 -> K=24..31
__device__ __forceinline__ v16h ldfragA(const _Float16* p, int lh) {
  union { v16h v; h8 h[2]; } u;
  u.h[0] = *(const h8*)(p + lh * 8);
  u.h[1] = *(const h8*)(p + 16 + lh * 8);
  return u.v;
}

// B fragment (32x16 f16, K x N): lane = half*16 + n; elems 0..7 -> K = half*8 + i, elems 8..15 -> K = 16 + half*8 + (i-8).
__device__ __forceinline__ v16h ldfragB(const _Float16* p, int lh) {
  union { v16h v; h8 h[2]; } u;
  u.h[0] = *(const h8*)(p + lh * 8);
  u.h[1] = *(const h8*)(p + 16 + lh * 8);
  return u.v;
}

__device__ __forceinline__ v8f wmma16(v16h a, v16h b, v8f c) {
  return __builtin_amdgcn_wmma_f32_16x16x32_f16(
      /*neg_a=*/false, a, /*neg_b=*/false, b,
      /*c_mod=*/(short)0, c, /*reuse_a=*/false, /*reuse_b=*/false);
}

// -------------------------- fused cosine GEMM ------------------------------
__global__ void __launch_bounds__(256)
cosgemm_kernel(const float* __restrict__ sup, const float* __restrict__ xh,
               const float* __restrict__ xnorm, const float* __restrict__ snorm,
               float* __restrict__ out) {
  __shared__ __align__(16) _Float16 Ah[BM * LDH];
  __shared__ __align__(16) _Float16 Al[BM * LDH];
  __shared__ __align__(16) _Float16 Bh[BN * LDH];
  __shared__ __align__(16) _Float16 Bl[BN * LDH];

  const int b   = blockIdx.z;
  const int tM  = blockIdx.y;          // tile index along T
  const int tN  = blockIdx.x;          // tile index along S
  const int tid = threadIdx.x;
  const int lane = tid & 31;
  const int wave = tid >> 5;
  const int wm = wave & 3;             // wave M slot: 0..3 (32 rows each)
  const int wn = wave >> 2;            // wave N slot: 0..1 (32 cols each)
  const int lr = lane & 15;            // row/col within 16
  const int lh = lane >> 4;            // lane half

  const float* Ag = xh  + ((size_t)b * kT + (size_t)tM * BM) * kD;
  const float* Bg = sup + ((size_t)b * kS + (size_t)tN * BN) * kD;

  const int ldRow = tid >> 3;          // 0..31
  const int ldCol = (tid & 7) * 4;     // k offset in elements (float4 granularity)

  v8f zero = {0.f, 0.f, 0.f, 0.f, 0.f, 0.f, 0.f, 0.f};
  v8f acc[2][2];
  #pragma unroll
  for (int i = 0; i < 2; ++i)
    #pragma unroll
    for (int j = 0; j < 2; ++j) acc[i][j] = zero;

  for (int k0 = 0; k0 < kD; k0 += BK) {
    // Stage A tile (128x32 fp32 -> hi/lo f16 in LDS); convert once per element.
    #pragma unroll
    for (int it = 0; it < 4; ++it) {
      const int r = ldRow + it * 32;
      f4 v = *(const f4*)(Ag + (size_t)r * kD + k0 + ldCol);
      h4 hi, lo;
      cvt4(v, hi, lo);
      *(h4*)&Ah[r * LDH + ldCol] = hi;
      *(h4*)&Al[r * LDH + ldCol] = lo;
    }
    // Stage B tile (64x32).
    #pragma unroll
    for (int it = 0; it < 2; ++it) {
      const int r = ldRow + it * 32;
      f4 v = *(const f4*)(Bg + (size_t)r * kD + k0 + ldCol);
      h4 hi, lo;
      cvt4(v, hi, lo);
      *(h4*)&Bh[r * LDH + ldCol] = hi;
      *(h4*)&Bl[r * LDH + ldCol] = lo;
    }
    __syncthreads();

    v16h a_h[2], a_l[2], b_h[2], b_l[2];
    #pragma unroll
    for (int sm = 0; sm < 2; ++sm) {
      const int row = (wm * 32 + sm * 16 + lr) * LDH;
      a_h[sm] = ldfragA(&Ah[row], lh);
      a_l[sm] = ldfragA(&Al[row], lh);
    }
    #pragma unroll
    for (int sn = 0; sn < 2; ++sn) {
      const int row = (wn * 32 + sn * 16 + lr) * LDH;
      b_h[sn] = ldfragB(&Bh[row], lh);
      b_l[sn] = ldfragB(&Bl[row], lh);
    }

    // Split-precision product: hi*hi + lo*hi + hi*lo (lo*lo ~ 2^-22, dropped).
    #pragma unroll
    for (int sm = 0; sm < 2; ++sm)
      #pragma unroll
      for (int sn = 0; sn < 2; ++sn) {
        acc[sm][sn] = wmma16(a_h[sm], b_h[sn], acc[sm][sn]);
        acc[sm][sn] = wmma16(a_l[sm], b_h[sn], acc[sm][sn]);
        acc[sm][sn] = wmma16(a_h[sm], b_l[sn], acc[sm][sn]);
      }
    __syncthreads();
  }

  // Epilogue: divide by max(nx*ns, eps), non-temporal stores (don't pollute L2,
  // inputs should stay resident for other tiles).
  const int tBase = tM * BM + wm * 32;
  const int sBase = tN * BN + wn * 32;
  #pragma unroll
  for (int sm = 0; sm < 2; ++sm) {
    const int t0 = tBase + sm * 16 + lh * 8;   // C layout: vgpr i -> M = i + 8*lh
    #pragma unroll
    for (int sn = 0; sn < 2; ++sn) {
      const int s = sBase + sn * 16 + lr;      // C layout: lane%16 -> N
      const float ns = snorm[b * kS + s];
      float* orow = out + ((size_t)b * kT + t0) * kS + s;
      #pragma unroll
      for (int i = 0; i < 8; ++i) {
        const float nx = xnorm[b * kT + t0 + i];
        const float denom = fmaxf(nx * ns, 1e-10f);
        __builtin_nontemporal_store(acc[sm][sn][i] / denom, orow + (size_t)i * kS);
      }
    }
  }
}

// -------------------------- launcher ---------------------------------------
extern "C" void kernel_launch(void* const* d_in, const int* in_sizes, int n_in,
                              void* d_out, int out_size, void* d_ws, size_t ws_size,
                              hipStream_t stream) {
  (void)in_sizes; (void)n_in; (void)out_size; (void)ws_size;
  const float* sup = (const float*)d_in[0];   // support_sets [B,S,D]
  const float* xh  = (const float*)d_in[1];   // X_hats       [B,T,D]
  float* out = (float*)d_out;                 // [B,T,S]

  float* xnorm = (float*)d_ws;                // B*T floats
  float* snorm = xnorm + kB * kT;             // B*S floats (64 KB total)

  row_norms_kernel<<<dim3(kB * (kT + kS)), 256, 0, stream>>>(sup, xh, xnorm, snorm);

  dim3 grid(kS / BN, kT / BM, kB);            // (16, 8, 8)
  cosgemm_kernel<<<grid, 256, 0, stream>>>(sup, xh, xnorm, snorm, out);
}
